// AttentionLayer_60936995996246
// MI455X (gfx1250) — matrix operand order repaired and checked by reading, not compiled, not run on hardware
//
#include <hip/hip_runtime.h>
#include <hip/hip_bf16.h>

// ---------------------------------------------------------------------------
// (B,T,S,C,E) = (16,1024,1024,1024,1024); every GEMM is 16384 x 1024 x 1024.
// ---------------------------------------------------------------------------
#define BB    16
#define TT    1024
#define DIM   1024
#define SCALE 0.70710678118654752f   // sqrt(0.5)

typedef __attribute__((ext_vector_type(16))) _Float16 v16h;
typedef __attribute__((ext_vector_type(8)))  _Float16 v8h;
typedef __attribute__((ext_vector_type(4)))  _Float16 v4h;
typedef __attribute__((ext_vector_type(2)))  _Float16 v2h;
typedef __attribute__((ext_vector_type(8)))  float    v8f;
typedef __attribute__((ext_vector_type(4)))  float    v4f;

// v_cvt_pk_rtz_f16_f32: pack two f32 into two f16 (builtin returns __fp16x2;
// bit-cast to our _Float16x2 type)
static __device__ __forceinline__ v2h pk2(float a, float b) {
    auto t = __builtin_amdgcn_cvt_pkrtz(a, b);
    return __builtin_bit_cast(v2h, t);
}

#define EPI_BIAS_RES 0   // (acc + bias[col] + add[row*DIM+col]) * SCALE
#define EPI_MASK     1   // mask[b*DIM+col] ? -inf : acc
#define EPI_SQRTS    2   // acc * s_sqrt[b]

// ---------------------------------------------------------------------------
// Tiled mixed-precision GEMM:  out[row,col] = sum_k A[row,k] * B(k,col)
// Block tile 128(M) x 128(N), K step 32. 8 waves; each wave owns 32x64 =
// 2x4 v_wmma_f32_16x16x32_f16 tiles. f32 globals are converted to f16 with
// packed cvt while staging into LDS; fragments are two ds_load_b128 each,
// matching the ISA 16-bit A/B VGPR layouts. Global loads for tile k+1 are
// prefetched into registers while tile k is consumed (software pipeline).
// ---------------------------------------------------------------------------
template<int EPI, bool B_KMAJOR>
__global__ __launch_bounds__(256)
void gemm_kernel(const float* __restrict__ A,
                 const float* __restrict__ Bm, long bStride,
                 const float* __restrict__ bias,
                 const float* __restrict__ add,
                 const int*   __restrict__ mask,
                 const float* __restrict__ s_sqrt,
                 float* __restrict__ out)
{
    constexpr int MT = 128, NT = 128, KT = 32, LDK = KT + 8;  // 80B row stride
    __shared__ __align__(16) _Float16 As[MT][LDK];
    __shared__ __align__(16) _Float16 Bs[NT][LDK];

    const int tid  = threadIdx.x;
    const int wid  = tid >> 5;
    const int lane = tid & 31;
    const int lp   = lane & 15;
    const int hi   = lane >> 4;
    const int wm   = wid & 3;        // 4 waves of 32 rows
    const int wn   = wid >> 2;       // 2 waves of 64 cols

    const int row0 = blockIdx.y * MT;
    const int col0 = blockIdx.x * NT;
    const int bi   = row0 / TT;
    const float* __restrict__ Bp = Bm + (size_t)bi * (size_t)bStride;

    // staging coordinates (literal trip counts so everything unrolls clean)
    const int sr = tid >> 3;          // 0..31 : row within 32-row slab
    const int sq = tid & 7;           // 0..7  : float4 column
    const int kn = tid & 127;         // k-major: n
    const int kk = tid >> 7;          // k-major: k-pair select (0/1)

    v4f pa[4];                        // A prefetch: 4 slabs of 32 rows
    v4f pbn[4];                       // B prefetch (n-major)
    float pbk[16];                    // B prefetch (k-major): 8 k-pairs

    auto loadA = [&](int k0) {
        #pragma unroll
        for (int it = 0; it < 4; ++it)
            pa[it] = *reinterpret_cast<const v4f*>(
                &A[(size_t)(row0 + it * 32 + sr) * DIM + k0 + (sq << 2)]);
    };
    auto loadB = [&](int k0) {
        if constexpr (B_KMAJOR) {
            #pragma unroll
            for (int it = 0; it < 8; ++it) {
                const int kp = it * 2 + kk;                    // 0..15
                const size_t g = (size_t)(k0 + kp * 2) * DIM + col0 + kn;
                pbk[it * 2 + 0] = Bp[g];
                pbk[it * 2 + 1] = Bp[g + DIM];
            }
        } else {
            #pragma unroll
            for (int it = 0; it < 4; ++it)
                pbn[it] = *reinterpret_cast<const v4f*>(
                    &Bp[(size_t)(col0 + it * 32 + sr) * DIM + k0 + (sq << 2)]);
        }
    };
    auto stage = [&]() {
        #pragma unroll
        for (int it = 0; it < 4; ++it) {
            const v2h l = pk2(pa[it].x, pa[it].y);
            const v2h h = pk2(pa[it].z, pa[it].w);
            v4h q; q[0] = l[0]; q[1] = l[1]; q[2] = h[0]; q[3] = h[1];
            *reinterpret_cast<v4h*>(&As[it * 32 + sr][sq << 2]) = q;
        }
        if constexpr (B_KMAJOR) {
            #pragma unroll
            for (int it = 0; it < 8; ++it) {
                const int kp = it * 2 + kk;
                const v2h p = pk2(pbk[it * 2], pbk[it * 2 + 1]);
                *reinterpret_cast<v2h*>(&Bs[kn][kp * 2]) = p;
            }
        } else {
            #pragma unroll
            for (int it = 0; it < 4; ++it) {
                const v2h l = pk2(pbn[it].x, pbn[it].y);
                const v2h h = pk2(pbn[it].z, pbn[it].w);
                v4h q; q[0] = l[0]; q[1] = l[1]; q[2] = h[0]; q[3] = h[1];
                *reinterpret_cast<v4h*>(&Bs[it * 32 + sr][sq << 2]) = q;
            }
        }
    };

    v8f acc[2][4] = {};

    loadA(0);
    loadB(0);
    for (int k0 = 0; k0 < DIM; k0 += KT) {
        stage();
        __syncthreads();
        if (k0 + KT < DIM) { loadA(k0 + KT); loadB(k0 + KT); }  // prefetch

        // fragments per ISA 16-bit layouts: two b128 LDS loads each
        v16h afr[2], bfr[4];
        #pragma unroll
        for (int sm = 0; sm < 2; ++sm) {
            const _Float16* ar = &As[wm * 32 + sm * 16 + lp][0];
            const v8h lo = *reinterpret_cast<const v8h*>(ar + hi * 8);
            const v8h hh = *reinterpret_cast<const v8h*>(ar + 16 + hi * 8);
            #pragma unroll
            for (int q = 0; q < 8; ++q) { afr[sm][q] = lo[q]; afr[sm][8 + q] = hh[q]; }
        }
        #pragma unroll
        for (int sn = 0; sn < 4; ++sn) {
            const _Float16* br = &Bs[wn * 64 + sn * 16 + lp][0];
            const v8h lo = *reinterpret_cast<const v8h*>(br + hi * 8);
            const v8h hh = *reinterpret_cast<const v8h*>(br + 16 + hi * 8);
            #pragma unroll
            for (int q = 0; q < 8; ++q) { bfr[sn][q] = lo[q]; bfr[sn][8 + q] = hh[q]; }
        }

        #pragma unroll
        for (int sm = 0; sm < 2; ++sm)
            #pragma unroll
            for (int sn = 0; sn < 4; ++sn)
                acc[sm][sn] = __builtin_amdgcn_wmma_f32_16x16x32_f16(
                    false, afr[sm], false, bfr[sn],
                    (short)0, acc[sm][sn], false, false);

        __syncthreads();
    }

    // epilogue + store (C/D layout: row = hi*8 + r, col = lane%16)
    #pragma unroll
    for (int sm = 0; sm < 2; ++sm)
        #pragma unroll
        for (int sn = 0; sn < 4; ++sn)
            #pragma unroll
            for (int r = 0; r < 8; ++r) {
                const int row = row0 + wm * 32 + sm * 16 + hi * 8 + r;
                const int col = col0 + wn * 64 + sn * 16 + lp;
                float v = acc[sm][sn][r];
                if constexpr (EPI == EPI_BIAS_RES) {
                    v = (v + bias[col] + add[(size_t)row * DIM + col]) * SCALE;
                } else if constexpr (EPI == EPI_MASK) {
                    if (mask[(size_t)bi * DIM + col] != 0) v = -__builtin_inff();
                } else { // EPI_SQRTS
                    v = v * s_sqrt[bi];
                }
                out[(size_t)row * DIM + col] = v;
            }
}

// ---------------------------------------------------------------------------
// Row softmax (one block per (b,t) row; 8 waves of 32, wave32 shuffles).
// ---------------------------------------------------------------------------
__global__ __launch_bounds__(256)
void softmax_kernel(const float* __restrict__ sc, float* __restrict__ attn)
{
    __shared__ float red[8];
    const int r   = blockIdx.x;
    const int tid = threadIdx.x;
    const int wid = tid >> 5, lane = tid & 31;
    const float* src = sc   + (size_t)r * DIM;
    float*       dst = attn + (size_t)r * DIM;

    float v[4];
    float m = -__builtin_inff();
    #pragma unroll
    for (int i = 0; i < 4; ++i) { v[i] = src[tid + 256 * i]; m = fmaxf(m, v[i]); }
    #pragma unroll
    for (int o = 16; o > 0; o >>= 1) m = fmaxf(m, __shfl_xor(m, o, 32));
    if (lane == 0) red[wid] = m;
    __syncthreads();
    float M = red[0];
    #pragma unroll
    for (int i = 1; i < 8; ++i) M = fmaxf(M, red[i]);
    __syncthreads();

    float s = 0.f;
    #pragma unroll
    for (int i = 0; i < 4; ++i) { v[i] = __expf(v[i] - M); s += v[i]; }
    #pragma unroll
    for (int o = 16; o > 0; o >>= 1) s += __shfl_xor(s, o, 32);
    if (lane == 0) red[wid] = s;
    __syncthreads();
    float S = 0.f;
    #pragma unroll
    for (int i = 0; i < 8; ++i) S += red[i];
    const float inv = 1.0f / S;
    #pragma unroll
    for (int i = 0; i < 4; ++i) dst[tid + 256 * i] = v[i] * inv;
}

// ---------------------------------------------------------------------------
// s_sqrt[b] = sqrt(S - sum(mask[b,:]))
// ---------------------------------------------------------------------------
__global__ __launch_bounds__(256)
void mask_count_kernel(const int* __restrict__ mask, float* __restrict__ s_sqrt)
{
    __shared__ int red[8];
    const int b = blockIdx.x, tid = threadIdx.x;
    const int wid = tid >> 5, lane = tid & 31;
    int c = 0;
    #pragma unroll
    for (int i = 0; i < 4; ++i) c += (mask[(size_t)b * DIM + tid + 256 * i] != 0);
    #pragma unroll
    for (int o = 16; o > 0; o >>= 1) c += __shfl_xor(c, o, 32);
    if (lane == 0) red[wid] = c;
    __syncthreads();
    if (tid == 0) {
        int tot = 0;
        #pragma unroll
        for (int i = 0; i < 8; ++i) tot += red[i];
        s_sqrt[b] = sqrtf((float)(DIM - tot));
    }
}

// ---------------------------------------------------------------------------
extern "C" void kernel_launch(void* const* d_in, const int* in_sizes, int n_in,
                              void* d_out, int out_size, void* d_ws, size_t ws_size,
                              hipStream_t stream)
{
    (void)in_sizes; (void)n_in; (void)out_size; (void)ws_size;

    const float* x     = (const float*)d_in[0];   // (B,T,C)
    const float* te    = (const float*)d_in[1];   // (B,T,E)
    const float* keys  = (const float*)d_in[2];   // (B,E,S)
    const float* vals  = (const float*)d_in[3];   // (B,S,E)
    const int*   msk   = (const int*)d_in[4];     // (B,S)
    const float* w_in  = (const float*)d_in[5];   // (E,C)
    const float* b_in  = (const float*)d_in[6];   // (E)
    const float* w_out = (const float*)d_in[7];   // (C,E)
    const float* b_out = (const float*)d_in[8];   // (C)

    float* out  = (float*)d_out;                          // (B,T,C)
    float* attn = out + (size_t)BB * TT * DIM;            // (B,T,S)

    float* ws = (float*)d_ws;
    float* H  = ws;                                       // (B*T, E)   64 MB
    float* SC = ws + (size_t)16 * 1024 * 1024;            // scores/ctx 64 MB
    float* SQ = ws + (size_t)32 * 1024 * 1024;            // 16 floats

    const long PB = (long)DIM * DIM;  // per-batch stride for keys / values
    const dim3 blk(256);
    const dim3 gg(DIM / 128, (BB * TT) / 128);            // (N tiles, M tiles)

    mask_count_kernel<<<BB, blk, 0, stream>>>(msk, SQ);

    // H = (x @ w_in^T + b_in + te) * sqrt(0.5)
    gemm_kernel<EPI_BIAS_RES, false><<<gg, blk, 0, stream>>>(
        x, w_in, 0L, b_in, te, nullptr, nullptr, H);

    // scores = H @ enc_keys, padding -> -inf
    gemm_kernel<EPI_MASK, true><<<gg, blk, 0, stream>>>(
        H, keys, PB, nullptr, nullptr, msk, nullptr, SC);

    // attn = softmax(scores)  -> second half of d_out
    softmax_kernel<<<BB * TT, blk, 0, stream>>>(SC, attn);

    // ctx = (attn @ enc_values) * sqrt(valid)   (reuse SC buffer)
    gemm_kernel<EPI_SQRTS, true><<<gg, blk, 0, stream>>>(
        attn, vals, PB, nullptr, nullptr, nullptr, SQ, SC);

    // out = (ctx @ w_out^T + b_out + x) * sqrt(0.5)
    gemm_kernel<EPI_BIAS_RES, false><<<gg, blk, 0, stream>>>(
        SC, w_out, 0L, b_out, x, nullptr, nullptr, out);
}
